// GlobalGraph_11149735101043
// MI455X (gfx1250) — matrix operand order repaired and checked by reading, not compiled, not run on hardware
//
#include <hip/hip_runtime.h>

#define BATCH 64
#define SEQ   1024
#define CDIM  128

typedef _Float16 v16h  __attribute__((ext_vector_type(16)));
typedef float    v8f   __attribute__((ext_vector_type(8)));
typedef unsigned int u32x4 __attribute__((ext_vector_type(4)));
typedef int      i32x8 __attribute__((ext_vector_type(8)));
typedef int      i32x4 __attribute__((ext_vector_type(4)));

#if defined(__has_builtin)
#  if __has_builtin(__builtin_amdgcn_tensor_load_to_lds)
#    define USE_TDM 1
#  endif
#endif
#ifndef USE_TDM
#  define USE_TDM 0
#endif

union frag16 { uint4 q[2]; v16h v; };

__device__ __forceinline__ v8f zero8() {
    v8f v;
#pragma unroll
    for (int i = 0; i < 8; ++i) v[i] = 0.0f;
    return v;
}

__device__ __forceinline__ v8f wmma_f16(v16h a, v16h b, v8f c) {
    // D = A(16x32 f16) x B(32x16 f16) + C(16x16 f32)
    return __builtin_amdgcn_wmma_f32_16x16x32_f16(
        false, a, false, b, (short)0, c, false, false);
}

// ---------------------------------------------------------------------------
// Kernel 1: QKV projection.  Q,K token-major f16; V stored TRANSPOSED per
// batch (Vt[batch][d][token]) so attention B fragments are contiguous.
// ---------------------------------------------------------------------------
__global__ __launch_bounds__(128) void qkv_proj_kernel(
    const float* __restrict__ x,
    const float* __restrict__ qw, const float* __restrict__ qb,
    const float* __restrict__ kw, const float* __restrict__ kb,
    const float* __restrict__ vw, const float* __restrict__ vb,
    _Float16* __restrict__ Q, _Float16* __restrict__ K, _Float16* __restrict__ Vt)
{
    const int which = blockIdx.y;
    const float* W  = (which == 0) ? qw : (which == 1) ? kw : vw;
    const float* Bv = (which == 0) ? qb : (which == 1) ? kb : vb;
    _Float16*   Out = (which == 0) ? Q  : (which == 1) ? K  : Vt;

    __shared__ __align__(16) _Float16 sW[CDIM * 136];
    __shared__ float sB[CDIM];

    for (int i = threadIdx.x; i < CDIM * CDIM; i += blockDim.x) {
        int r = i >> 7, c = i & 127;
        sW[r * 136 + c] = (_Float16)W[i];
    }
    if (threadIdx.x < CDIM) sB[threadIdx.x] = Bv[threadIdx.x];
    __syncthreads();

    const int wave = threadIdx.x >> 5;
    const int lane = threadIdx.x & 31;
    const int lg   = lane >> 4;
    const int ll   = lane & 15;

    const int row0 = blockIdx.x * 64 + wave * 16;
    const float* xr = x + (size_t)(row0 + ll) * CDIM;

    v16h a[4];
#pragma unroll
    for (int kc = 0; kc < 4; ++kc) {
        int base = kc * 32 + lg * 8;
#pragma unroll
        for (int h = 0; h < 8; ++h) {
            a[kc][h]     = (_Float16)xr[base + h];
            a[kc][h + 8] = (_Float16)xr[base + 16 + h];
        }
    }

#pragma unroll
    for (int jt = 0; jt < 8; ++jt) {
        const int j = jt * 16 + ll;
        v8f c = zero8();
#pragma unroll
        for (int kc = 0; kc < 4; ++kc) {
            frag16 b;
            const _Float16* bp = &sW[j * 136 + kc * 32 + lg * 8];
            b.q[0] = *(const uint4*)(bp);
            b.q[1] = *(const uint4*)(bp + 16);
            c = wmma_f16(a[kc], b.v, c);
        }
        const float bias = sB[j];
#pragma unroll
        for (int r = 0; r < 8; ++r) {
            const int orow = row0 + r + lg * 8;
            const _Float16 h = (_Float16)(c[r] + bias);
            if (which == 2) {
                Out[((size_t)(orow >> 10) << 17) + ((size_t)j << 10) + (orow & 1023)] = h;
            } else {
                Out[(size_t)orow * CDIM + j] = h;
            }
        }
    }
}

// ---------------------------------------------------------------------------
// Kernel 2: fused masked flash-attention, S^T formulation.
// K chunks: global_load_async_to_lds_b128 (ASYNCcnt).
// Vt chunks: Tensor Data Mover tensor_load_to_lds with LDS padding
//            (TENSORcnt), one descriptor per block iteration.
// ---------------------------------------------------------------------------
__global__ __launch_bounds__(256) void attn_kernel(
    const _Float16* __restrict__ Q, const _Float16* __restrict__ K,
    const _Float16* __restrict__ Vt, const int* __restrict__ valid_lens,
    float* __restrict__ out)
{
    __shared__ __align__(16) _Float16 sK[32 * 136];    // K chunk, padded rows
#if USE_TDM
    __shared__ __align__(16) _Float16 sV[CDIM * 40];   // Vt chunk: 128 rows x 32 keys, pad 8
#endif

    const int batch = blockIdx.x >> 3;
    const int qt    = blockIdx.x & 7;
    const int wave  = threadIdx.x >> 5;
    const int lane  = threadIdx.x & 31;
    const int lg    = lane >> 4, ll = lane & 15;
    const int vl    = valid_lens[batch];

    const int    qrow_local = qt * 128 + wave * 16;
    const size_t bbase      = (size_t)batch * SEQ * CDIM;
    const size_t vtbase     = (size_t)batch * CDIM * SEQ;

    // Q as B fragments: lane = query (ll), halves = feature dim d
    v16h qb[4];
    {
        const _Float16* qr = Q + bbase + (size_t)(qrow_local + ll) * CDIM;
#pragma unroll
        for (int kc = 0; kc < 4; ++kc) {
            frag16 f;
            const _Float16* p = qr + kc * 32 + lg * 8;
            f.q[0] = *(const uint4*)(p);
            f.q[1] = *(const uint4*)(p + 16);
            qb[kc] = f.v;
        }
    }

    const int  qvalid = (qrow_local + ll) < vl;
    float m_q = -3.0e38f, l_q = 0.0f;
    v8f o[8];
#pragma unroll
    for (int jt = 0; jt < 8; ++jt) o[jt] = zero8();

    const uint64_t kbase = (uint64_t)(uintptr_t)(K + bbase);
#if USE_TDM
    const uint64_t vbase64 = (uint64_t)(uintptr_t)(Vt + vtbase);
    const unsigned sv_lds  = (unsigned)(uintptr_t)&sV[0];
#endif

    for (int mt = 0; mt < SEQ / 32; ++mt) {
        __syncthreads();
        // ---- async staged load: K chunk 32 x 128 f16 (two b128 per thread)
#pragma unroll
        for (int u = 0; u < 2; ++u) {
            const int m   = (threadIdx.x >> 4) + u * 16;
            const int seg = threadIdx.x & 15;
            const unsigned ldsa = (unsigned)(uintptr_t)&sK[m * 136 + seg * 8];
            const unsigned voff = (unsigned)((((mt * 32 + m) * CDIM) + seg * 8) * 2);
            asm volatile("global_load_async_to_lds_b128 %0, %1, %2"
                         :: "v"(ldsa), "v"(voff), "s"(kbase) : "memory");
        }
#if USE_TDM
        // ---- TDM: Vt tile 128 rows x 32 keys -> LDS, padded stride 40 halfs
        if (wave == 0) {
            const uint64_t gaddr = vbase64 + (uint64_t)mt * 64;   // tile start (bytes)
            u32x4 g0 = { 1u,                                      // count=1, user desc
                         sv_lds,                                  // lds_addr
                         (unsigned)gaddr,
                         (unsigned)(gaddr >> 32) | (2u << 30) };  // addr[56:32] | type=2
            i32x8 g1 = { (int)((1u << 16)     // data_size = 2B
                             | (1u << 20)     // pad_enable
                             | (3u << 22)     // pad_interval: 16 DWORDs (64B rows)
                             | (3u << 25)),   // pad_amount: 4 DWORDs (8 halfs)
                         (int)((SEQ & 0xFFFF) << 16),   // tensor_dim0 = 1024 (lo16)
                         (int)((CDIM & 0xFFFF) << 16),  // dim0 hi=0 | tensor_dim1 = 128
                         (int)(32u << 16),              // dim1 hi=0 | tile_dim0 = 32
                         (int)CDIM,                     // tile_dim1 = 128, tile_dim2 = 0
                         (int)SEQ,                      // tensor_dim0_stride = 1024 (lo32)
                         0, 0 };
            i32x4 z4 = { 0, 0, 0, 0 };
#if __clang_major__ >= 23
            i32x8 z8 = { 0, 0, 0, 0, 0, 0, 0, 0 };
            __builtin_amdgcn_tensor_load_to_lds(g0, g1, z4, z4, z8, 0);
#else
            __builtin_amdgcn_tensor_load_to_lds(g0, g1, z4, z4, 0);
#endif
            __builtin_amdgcn_s_wait_tensorcnt(0);
        }
#endif
        if (mt + 1 < SEQ / 32) {
            __builtin_prefetch(Vt + vtbase + ((size_t)(threadIdx.x & 127) << 10)
                                   + (mt + 1) * 32, 0, 0);
        }
        asm volatile("s_wait_asynccnt 0" ::: "memory");
        __syncthreads();

        // ---- S^T tiles (keys x queries): A = K rows from LDS, B = Q^T regs
        v8f s0 = zero8(), s1 = zero8();
#pragma unroll
        for (int kc = 0; kc < 4; ++kc) {
            frag16 a0, a1;
            const _Float16* a0p = &sK[(ll)      * 136 + kc * 32 + lg * 8];
            const _Float16* a1p = &sK[(16 + ll) * 136 + kc * 32 + lg * 8];
            a0.q[0] = *(const uint4*)(a0p);
            a0.q[1] = *(const uint4*)(a0p + 16);
            a1.q[0] = *(const uint4*)(a1p);
            a1.q[1] = *(const uint4*)(a1p + 16);
            s0 = wmma_f16(a0.v, qb[kc], s0);
            s1 = wmma_f16(a1.v, qb[kc], s1);
        }

        // ---- per-query (= per-lane) online softmax across the 32 new keys
        const int key0 = mt * 32 + lg * 8;
        float sv0[8], sv1[8];
        float tmax = -3.0e38f;
#pragma unroll
        for (int r = 0; r < 8; ++r) {
            sv0[r] = (qvalid && (key0 + r)      < vl) ? s0[r] : -3.0e38f;
            sv1[r] = (qvalid && (key0 + 16 + r) < vl) ? s1[r] : -3.0e38f;
            tmax = fmaxf(tmax, fmaxf(sv0[r], sv1[r]));
        }
        tmax = fmaxf(tmax, __shfl_xor(tmax, 16, 32));
        const float mn    = fmaxf(m_q, tmax);
        const float alpha = __expf(m_q - mn);
        m_q = mn;

        v16h pa;                 // P^T registers ARE the next A fragment
        float rs = 0.0f;
#pragma unroll
        for (int r = 0; r < 8; ++r) {
            const float e0 = (sv0[r] > -1.0e38f) ? __expf(sv0[r] - mn) : 0.0f;
            const float e1 = (sv1[r] > -1.0e38f) ? __expf(sv1[r] - mn) : 0.0f;
            rs += e0 + e1;
            pa[r]     = (_Float16)e0;
            pa[r + 8] = (_Float16)e1;
        }
        rs += __shfl_xor(rs, 16, 32);
        l_q = l_q * alpha + rs;

#pragma unroll
        for (int r = 0; r < 8; ++r) {
            const float ar = __shfl(alpha, r + lg * 8, 16);
#pragma unroll
            for (int jt = 0; jt < 8; ++jt) o[jt][r] *= ar;
        }

        // ---- O += P x V
#if USE_TDM
#pragma unroll
        for (int jt = 0; jt < 8; ++jt) {
            frag16 vf;
            const _Float16* vp = &sV[(jt * 16 + ll) * 40 + lg * 8];
            vf.q[0] = *(const uint4*)(vp);
            vf.q[1] = *(const uint4*)(vp + 16);
            o[jt] = wmma_f16(pa, vf.v, o[jt]);
        }
#else
        const _Float16* vchunk = Vt + vtbase + (size_t)(mt * 32 + lg * 8);
#pragma unroll
        for (int jt = 0; jt < 8; ++jt) {
            frag16 vf;
            const _Float16* vp = vchunk + ((size_t)(jt * 16 + ll) << 10);
            vf.q[0] = *(const uint4*)(vp);
            vf.q[1] = *(const uint4*)(vp + 16);
            o[jt] = wmma_f16(pa, vf.v, o[jt]);
        }
#endif
    }

    // ---- normalize and store
#pragma unroll
    for (int r = 0; r < 8; ++r) {
        const float denom = __shfl(l_q, r + lg * 8, 16);
        const float inv   = (denom > 0.0f) ? (1.0f / denom) : 0.0f;
        const int   row   = qrow_local + r + lg * 8;
#pragma unroll
        for (int jt = 0; jt < 8; ++jt) {
            out[bbase + (size_t)row * CDIM + jt * 16 + ll] = o[jt][r] * inv;
        }
    }
}

// ---------------------------------------------------------------------------
extern "C" void kernel_launch(void* const* d_in, const int* in_sizes, int n_in,
                              void* d_out, int out_size, void* d_ws, size_t ws_size,
                              hipStream_t stream) {
    const float* x    = (const float*)d_in[0];
    const int*   vlen = (const int*)d_in[1];
    const float* qw   = (const float*)d_in[2];
    const float* qb   = (const float*)d_in[3];
    const float* kw   = (const float*)d_in[4];
    const float* kb   = (const float*)d_in[5];
    const float* vw   = (const float*)d_in[6];
    const float* vb   = (const float*)d_in[7];
    float*       out  = (float*)d_out;

    const size_t elems = (size_t)BATCH * SEQ * CDIM;
    _Float16* Q  = (_Float16*)d_ws;
    _Float16* K  = Q + elems;
    _Float16* Vt = K + elems;

    qkv_proj_kernel<<<dim3(BATCH * SEQ / 64, 3), 128, 0, stream>>>(
        x, qw, qb, kw, kb, vw, vb, Q, K, Vt);
    attn_kernel<<<dim3(BATCH * (SEQ / 128)), 256, 0, stream>>>(
        Q, K, Vt, vlen, out);
}
